// ScanModel_22076131901868
// MI455X (gfx1250) — matrix operand order repaired and checked by reading, not compiled, not run on hardware
//
#include <hip/hip_runtime.h>
#include <hip/hip_bf16.h>

// ---------------------------------------------------------------------------
// LIF scan model, restructured for MI455X:
//   Phase 1: cur = X @ W_in     (256000x700)@(700x100)  -- bf16 WMMA, HBM-bound
//   Phase 2: sequential LIF update, elementwise only (25600 parallel chains)
//   Phase 3: out = Z @ W_out    (256000x100)@(100x10)   -- bf16 WMMA
// ---------------------------------------------------------------------------

typedef __attribute__((ext_vector_type(16))) __bf16 bf16x16;
typedef __attribute__((ext_vector_type(8)))  float  f32x8;
typedef __attribute__((ext_vector_type(4)))  float  f32x4;
typedef __attribute__((ext_vector_type(2)))  unsigned int u32x2;

union FragAB { bf16x16 v; uint4 q[2]; };

// Problem constants
constexpr int NB   = 256;
constexpr int T    = 1000;
constexpr int DIN  = 700;
constexpr int HID  = 100;
constexpr int NOUT = 10;
constexpr int M1   = NB * T;        // 256000 GEMM rows
constexpr int KP1  = 704;           // K padded to multiple of 32
constexpr int NP1  = 112;           // N padded to 7 WMMA tiles
constexpr int KC   = 32;            // K per WMMA
constexpr int NKIT = KP1 / KC;      // 22
constexpr int BM   = 128;           // M rows per workgroup (8 waves x 16)
constexpr int HIDP = 128;           // hidden padded for phase-3 K

__device__ __forceinline__ unsigned short f32_to_bf16_rte(float f) {
  unsigned int u = __float_as_uint(f);
  u += 0x7FFFu + ((u >> 16) & 1u);   // round-to-nearest-even
  return (unsigned short)(u >> 16);
}

// ---------------------------------------------------------------------------
// Kernel 0: convert + pad weights to bf16, transposed (n-major) so WMMA B
// fragments are contiguous 16-K runs per lane.
//   Bp    : [NP1][KP1]  bf16, Bp[n][k] = W_in[k][n]   (zero padded)
//   Woutp : [16][HIDP]  bf16, Woutp[n][k] = W_out[k][n] (zero padded)
// ---------------------------------------------------------------------------
__global__ void convert_weights(const float* __restrict__ Win,
                                const float* __restrict__ Wout,
                                unsigned short* __restrict__ Bp,
                                unsigned short* __restrict__ Woutp) {
  int idx = blockIdx.x * blockDim.x + threadIdx.x;
  const int nBp = NP1 * KP1;
  if (idx < nBp) {
    int nn = idx / KP1, k = idx % KP1;
    unsigned short r = 0;
    if (nn < HID && k < DIN) r = f32_to_bf16_rte(Win[k * HID + nn]);
    Bp[idx] = r;
  } else {
    int j = idx - nBp;
    if (j < 16 * HIDP) {
      int nn = j / HIDP, k = j % HIDP;
      unsigned short r = 0;
      if (nn < NOUT && k < HID) r = f32_to_bf16_rte(Wout[k * NOUT + nn]);
      Woutp[j] = r;
    }
  }
}

// ---------------------------------------------------------------------------
// Phase-1 GEMM helpers: global->register chunk fetch and register->LDS stage.
// Per thread per K-chunk: 4x b128 nontemporal loads of X, 7 dword loads of B;
// then 4x ds_store_b64 (packed bf16) + 7 dword LDS stores.
// ---------------------------------------------------------------------------
__device__ __forceinline__ void g1_load(const float* __restrict__ X,
                                        const unsigned int* __restrict__ Bq,
                                        long mbase, int tid, int kb,
                                        f32x4 af[4], unsigned int bw[7]) {
#pragma unroll
  for (int i = 0; i < 4; ++i) {
    int idx = i * 256 + tid;            // 0..1023 float4 slots (128 rows x 8)
    int r   = idx >> 3;
    int k4  = kb + ((idx & 7) << 2);    // multiple of 4; DIN is too
    if (k4 < DIN) {
      af[i] = __builtin_nontemporal_load((const f32x4*)&X[(mbase + r) * DIN + k4]);
    } else {
      f32x4 z = {0.f, 0.f, 0.f, 0.f};
      af[i] = z;                        // zero K padding (700..703)
    }
  }
#pragma unroll
  for (int i = 0; i < 7; ++i) {
    int idx = i * 256 + tid;            // 0..1791 dwords (112 cols x 16)
    int nn = idx >> 4, kd = idx & 15;
    bw[i] = Bq[nn * (KP1 / 2) + (kb >> 1) + kd];
  }
}

__device__ __forceinline__ void g1_store(unsigned short* __restrict__ Abuf,
                                         unsigned short* __restrict__ Bbuf,
                                         int tid, const f32x4 af[4],
                                         const unsigned int bw[7]) {
  unsigned int* As32 = (unsigned int*)Abuf;
#pragma unroll
  for (int i = 0; i < 4; ++i) {
    int idx = i * 256 + tid;
    int r = idx >> 3, kq = (idx & 7) << 1;  // dword index within 16-dword row
    u32x2 d;
    d.x = (unsigned int)f32_to_bf16_rte(af[i].x) |
          ((unsigned int)f32_to_bf16_rte(af[i].y) << 16);
    d.y = (unsigned int)f32_to_bf16_rte(af[i].z) |
          ((unsigned int)f32_to_bf16_rte(af[i].w) << 16);
    *(u32x2*)&As32[r * 16 + kq] = d;        // ds_store_b64
  }
  unsigned int* Bs32 = (unsigned int*)Bbuf;
#pragma unroll
  for (int i = 0; i < 7; ++i) Bs32[i * 256 + tid] = bw[i];
}

// ---------------------------------------------------------------------------
// Kernel 1: cur[M1][HID] = X[M1][DIN] @ W_in[DIN][HID], bf16 WMMA.
// 256 threads = 8 waves; wave w owns M rows [w*16, w*16+16), all 7 N tiles.
// Double-buffered LDS, one barrier per K step; next chunk's global loads are
// issued before the WMMA block so they overlap matrix math.
// ---------------------------------------------------------------------------
__global__ __launch_bounds__(256) void gemm1_wmma(
    const float* __restrict__ X,
    const unsigned short* __restrict__ Bp,
    float* __restrict__ cur) {
  __shared__ __align__(16) unsigned short Asub[2][BM * KC];   // 2 x 8 KB
  __shared__ __align__(16) unsigned short Bsub[2][NP1 * KC];  // 2 x 7 KB

  const int tid  = threadIdx.x;
  const int wave = tid >> 5;
  const int lane = tid & 31;
  const int half = lane >> 4;
  const int l16  = lane & 15;
  const long mbase = (long)blockIdx.x * BM;
  const unsigned int* Bq = (const unsigned int*)Bp;

  f32x8 acc[7];
#pragma unroll
  for (int nt = 0; nt < 7; ++nt)
#pragma unroll
    for (int r = 0; r < 8; ++r) acc[nt][r] = 0.0f;

  f32x4 af[4];
  unsigned int bw[7];

  // Prologue: stage chunk 0 into buffer 0
  g1_load(X, Bq, mbase, tid, 0, af, bw);
  g1_store(Asub[0], Bsub[0], tid, af, bw);
  __syncthreads();

  const int arow = wave * 16 + l16;
  const int ko   = half * 16;

  int p = 0;
  for (int it = 0; it < NKIT; ++it) {
    const bool more = (it + 1 < NKIT);
    if (more) g1_load(X, Bq, mbase, tid, (it + 1) * KC, af, bw);

    // A fragment: lanes 0-15 -> K 0..15; lanes 16-31 -> K 16..31
    FragAB a;
    a.q[0] = *(const uint4*)&Asub[p][arow * KC + ko];
    a.q[1] = *(const uint4*)&Asub[p][arow * KC + ko + 8];

#pragma unroll
    for (int nt = 0; nt < 7; ++nt) {
      // B fragment: lanes 0-15 -> K 0..15; lanes 16-31 -> K 16..31
      const int col = nt * 16 + l16;
      FragAB b;
      b.q[0] = *(const uint4*)&Bsub[p][col * KC + half * 16];
      b.q[1] = *(const uint4*)&Bsub[p][col * KC + half * 16 + 8];
      acc[nt] = __builtin_amdgcn_wmma_f32_16x16x32_bf16(
          false, a.v, false, b.v, (short)0, acc[nt], false, false);
    }

    if (more) g1_store(Asub[1 - p], Bsub[1 - p], tid, af, bw);
    __syncthreads();
    p ^= 1;
  }

  // Epilogue: C layout -> VGPR r is row r (+8 for upper lane half), N = lane&15
  const long mrow = mbase + wave * 16 + half * 8;
#pragma unroll
  for (int nt = 0; nt < 7; ++nt) {
    int col = nt * 16 + l16;
    if (col < HID) {
#pragma unroll
      for (int r = 0; r < 8; ++r)
        cur[(mrow + r) * HID + col] = acc[nt][r];
    }
  }
}

// ---------------------------------------------------------------------------
// Kernel 2: sequential LIF scan. Block = one sample, thread = one hidden unit.
// Writes spikes as bf16 {0,1} into zbuf[M1][HIDP] (K padded to 128 with zeros).
// ---------------------------------------------------------------------------
__global__ __launch_bounds__(128) void lif_scan(
    const float* __restrict__ cur,
    const float* __restrict__ state0,
    unsigned short* __restrict__ zbuf) {
  const int n = blockIdx.x;
  const int h = threadIdx.x;
  unsigned short* zp = zbuf + (long)n * T * HIDP + h;
  if (h < HID) {
    float iC = state0[0 * NB * HID + n * HID + h];
    float v  = state0[1 * NB * HID + n * HID + h];
    float z  = state0[2 * NB * HID + n * HID + h];
    const float* c = cur + (long)n * T * HID + h;
    for (int t = 0; t < T; ++t) {
      iC = 0.9f * iC + c[(long)t * HID];
      v  = 0.9f * v * (1.0f - z) + iC;
      bool spk = (v - 1.0f) > 0.0f;
      z = spk ? 1.0f : 0.0f;
      zp[(long)t * HIDP] = spk ? (unsigned short)0x3F80u : (unsigned short)0u;
    }
  } else {
    for (int t = 0; t < T; ++t) zp[(long)t * HIDP] = 0u;  // zero K padding
  }
}

// ---------------------------------------------------------------------------
// Kernel 3: out[M1][NOUT] = Z[M1][HIDP] @ W_out. One WMMA N tile, K=128 in 4
// chunks. One wave per 16-row M tile (2000 blocks x 8 waves = 16000 tiles).
// ---------------------------------------------------------------------------
__global__ __launch_bounds__(256) void gemm2_wmma(
    const unsigned short* __restrict__ zbuf,
    const unsigned short* __restrict__ Woutp,
    float* __restrict__ out) {
  const int tid  = threadIdx.x;
  const int wave = tid >> 5;
  const int lane = tid & 31;
  const int half = lane >> 4;
  const int l16  = lane & 15;
  const long tile  = (long)blockIdx.x * 8 + wave;
  const long mbase = tile * 16;
  const int ko = half * 16;

  f32x8 acc;
#pragma unroll
  for (int r = 0; r < 8; ++r) acc[r] = 0.0f;

#pragma unroll
  for (int c = 0; c < 4; ++c) {
    FragAB a, b;
    const unsigned short* ap = zbuf + (mbase + l16) * HIDP + c * 32 + ko;
    a.q[0] = *(const uint4*)ap;
    a.q[1] = *(const uint4*)(ap + 8);
    const unsigned short* bp = Woutp + l16 * HIDP + c * 32 + half * 16;
    b.q[0] = *(const uint4*)bp;
    b.q[1] = *(const uint4*)(bp + 8);
    acc = __builtin_amdgcn_wmma_f32_16x16x32_bf16(
        false, a.v, false, b.v, (short)0, acc, false, false);
  }

  if (l16 < NOUT) {
    const long mrow = mbase + half * 8;
#pragma unroll
    for (int r = 0; r < 8; ++r)
      out[(mrow + r) * NOUT + l16] = acc[r];
  }
}

// ---------------------------------------------------------------------------
extern "C" void kernel_launch(void* const* d_in, const int* in_sizes, int n_in,
                              void* d_out, int out_size, void* d_ws, size_t ws_size,
                              hipStream_t stream) {
  const float* x      = (const float*)d_in[0];   // (256,1000,700)
  const float* state0 = (const float*)d_in[1];   // (3,256,100)
  const float* Win    = (const float*)d_in[2];   // (700,100)
  const float* Wout   = (const float*)d_in[3];   // (100,10)
  float* out = (float*)d_out;                    // (256,1000,10)

  // Workspace layout (all offsets 256B-aligned):
  //   Bp    : 112*704*2      = 157,696 B
  //   Woutp : 16*128*2       =   4,096 B
  //   cur   : 256000*100*4   = 102,400,000 B
  //   zbuf  : 256000*128*2   =  65,536,000 B
  char* ws = (char*)d_ws;
  unsigned short* Bp    = (unsigned short*)(ws);
  unsigned short* Woutp = (unsigned short*)(ws + 157696);
  float*          cur   = (float*)(ws + 157696 + 4096);
  unsigned short* zbuf  = (unsigned short*)(ws + 157696 + 4096 + (size_t)M1 * HID * 4);

  convert_weights<<<(NP1 * KP1 + 16 * HIDP + 255) / 256, 256, 0, stream>>>(
      Win, Wout, Bp, Woutp);
  gemm1_wmma<<<M1 / BM, 256, 0, stream>>>(x, Bp, cur);
  lif_scan<<<NB, 128, 0, stream>>>(cur, state0, zbuf);
  gemm2_wmma<<<(M1 / 16) / 8, 256, 0, stream>>>(zbuf, Woutp, out);
}
